// PackedTernaryLinear_18614388260896
// MI455X (gfx1250) — matrix operand order repaired and checked by reading, not compiled, not run on hardware
//
#include <hip/hip_runtime.h>
#include <hip/hip_bf16.h>

// y[t,o] = sum_i x[t,i] * (alpha * W[o,i]) + bias[o]
// W ternary packed 4/byte LSB-first: 0->0, 1->+1, 2->-1, 3->0.
//
// Pass 1: x fp32 -> bf16 into d_ws (once; bandwidth-trivial at 23.3 TB/s).
// Pass 2: bf16 WMMA GEMM, 128x128 tile/block, K-step 64, 8 waves.
//         x tile staged into LDS by the Tensor Data Mover (tensor_load_to_lds,
//         TENSORcnt), weights decoded 2-bit -> bf16 by VALU, double-buffered.

typedef __bf16 bf16x16 __attribute__((ext_vector_type(16)));
typedef __bf16 bf16x8  __attribute__((ext_vector_type(8)));
typedef float  f32x8   __attribute__((ext_vector_type(8)));
typedef unsigned int v4u __attribute__((ext_vector_type(4)));
typedef int v8i __attribute__((ext_vector_type(8)));
typedef int v4i __attribute__((ext_vector_type(4)));

#define OUT_F 4096
#define IN_F  4096
#define NTOK  4096
#define BM 128
#define BN 128
#define BK 64

// ---------------- pass 1: x fp32 -> bf16 (RNE) ----------------
__global__ void __launch_bounds__(256)
x_to_bf16_kernel(const float* __restrict__ x, unsigned short* __restrict__ xb)
{
    const size_t i = ((size_t)blockIdx.x * 256 + threadIdx.x) * 16;
    const float4* xp = (const float4*)(x + i);
    float4 f0 = xp[0], f1 = xp[1], f2 = xp[2], f3 = xp[3];
    float fv[16];
    fv[0]=f0.x; fv[1]=f0.y; fv[2]=f0.z; fv[3]=f0.w;
    fv[4]=f1.x; fv[5]=f1.y; fv[6]=f1.z; fv[7]=f1.w;
    fv[8]=f2.x; fv[9]=f2.y; fv[10]=f2.z; fv[11]=f2.w;
    fv[12]=f3.x; fv[13]=f3.y; fv[14]=f3.z; fv[15]=f3.w;
    unsigned w[8];
#pragma unroll
    for (int j = 0; j < 8; ++j) {
        unsigned lo = __float_as_uint(fv[2 * j]);
        unsigned hi = __float_as_uint(fv[2 * j + 1]);
        lo = (lo + 0x7FFFu + ((lo >> 16) & 1u)) >> 16;
        hi = (hi + 0x7FFFu + ((hi >> 16) & 1u)) >> 16;
        w[j] = lo | (hi << 16);
    }
    uint4* dst = (uint4*)(xb + i);
    dst[0] = make_uint4(w[0], w[1], w[2], w[3]);
    dst[1] = make_uint4(w[4], w[5], w[6], w[7]);
}

// ---------------- TDM: 2D tile (BK x BM rows) of bf16 into LDS ----------------
// D# per CDNA5 ISA ch.8: group0 = {count/flags, lds_addr, ga_lo, ga_hi|type},
// group1 = {ds/flags, dims, tiles, stride}, groups 2/3/4 zero (2D tensor).
__device__ __forceinline__ void tdm_load_x_tile(const unsigned short* src,
                                                unsigned lds_addr)
{
    unsigned long long ga = (unsigned long long)(size_t)src;
    v4u g0 = { 1u,                                   // count=1, user mode
               lds_addr,                             // LDS byte address
               (unsigned)(ga & 0xFFFFFFFFu),         // global_addr[31:0]
               (unsigned)(((ga >> 32) & 0x1FFFFFFu)  // global_addr[56:32]
                          | (2u << 30)) };           // type=2 (image)
    v8i g1 = { (int)(1u << 16),                      // data_size=1 (2 bytes)
               (int)((IN_F & 0xFFFFu) << 16),        // tensor_dim0 lo16 @63:48
               (int)(((IN_F >> 16) & 0xFFFFu)        // tensor_dim0 hi16
                     | ((NTOK & 0xFFFFu) << 16)),    // tensor_dim1 lo16
               (int)(((NTOK >> 16) & 0xFFFFu)
                     | (BK << 16)),                  // tile_dim0 = 64
               (int)(BM),                            // tile_dim1=128, tile_dim2=0
               (int)(IN_F),                          // tensor_dim0_stride lo32
               0, 0 };
    v4i z4 = { 0, 0, 0, 0 };
    v8i z8 = { 0, 0, 0, 0, 0, 0, 0, 0 };
    __builtin_amdgcn_tensor_load_to_lds(g0, g1, z4, z4, z8, 0);
}

// ---------------- pass 2: GEMM ----------------
__global__ void __launch_bounds__(256)
ternary_wmma_bf16_kernel(const unsigned short* __restrict__ xb,
                         const unsigned char* __restrict__ packed,
                         const float* __restrict__ alpha_p,
                         const float* __restrict__ bias,
                         float* __restrict__ out)
{
    // double-buffered tiles: Xs [buf][m][k] bf16, Ws [buf][n][k] bf16 (as u32)
    __shared__ __align__(128) unsigned short Xs[2][BM * BK];     // 2 x 16 KB
    __shared__ __align__(16)  unsigned int   Ws[2][BN * BK / 2]; // 2 x 16 KB

    const int tid  = threadIdx.x;
    const int lane = tid & 31;
    const int wave = tid >> 5;

    const int m0 = blockIdx.y * BM;
    const int n0 = blockIdx.x * BN;

    const int wm = wave >> 1;   // 0..3 : 32-row M slab
    const int wn = wave & 1;    // 0..1 : 64-col N slab

    const int row  = tid >> 1;  // 0..127 (weight-decode row = n)
    const int half = tid & 1;   // 32-element K half of the 64-wide tile

    const int h  = lane >> 4;
    const int ml = lane & 15;
    const int kb = h * 16;

    f32x8 acc[2][4] = {};

    // ---- decode one u64 of codes (32 weights) into LDS buffer `buf` ----
    auto stage_w = [&](int kk, int buf) {
        const uint2 wb = *(const uint2*)(
            packed + (size_t)(n0 + row) * (IN_F / 4) + (kk >> 2) + half * 8);
        unsigned w[16];
#pragma unroll
        for (int p = 0; p < 2; ++p) {
            const unsigned wbits = p ? wb.y : wb.x;
#pragma unroll
            for (int j = 0; j < 8; ++j) {
                unsigned c  = wbits >> (4 * j);
                unsigned b0a = c & 1u,        b1a = (c >> 1) & 1u;
                unsigned b0b = (c >> 2) & 1u, b1b = (c >> 3) & 1u;
                unsigned h0 = ((b0a ^ b1a) * 0x3F80u) | (b1a << 15);
                unsigned h1 = ((b0b ^ b1b) * 0x3F80u) | (b1b << 15);
                w[p * 8 + j] = h0 | (h1 << 16);
            }
        }
        uint4* dst = (uint4*)&Ws[buf][row * 32 + half * 16];
        dst[0] = make_uint4(w[0],  w[1],  w[2],  w[3]);
        dst[1] = make_uint4(w[4],  w[5],  w[6],  w[7]);
        dst[2] = make_uint4(w[8],  w[9],  w[10], w[11]);
        dst[3] = make_uint4(w[12], w[13], w[14], w[15]);
    };

    // ---- prologue: tile 0 into buffer 0 ----
    stage_w(0, 0);
    if (wave == 0) {
        tdm_load_x_tile(xb + (size_t)m0 * IN_F, (unsigned)(size_t)&Xs[0][0]);
        __builtin_amdgcn_s_wait_tensorcnt(0);
    }
    __syncthreads();

    int cur = 0;
    for (int kk = 0; kk < IN_F; kk += BK) {
        const int nxt = cur ^ 1;

        // stage next tile while computing current one
        if (kk + BK < IN_F) {
            stage_w(kk + BK, nxt);
            if (wave == 0)
                tdm_load_x_tile(xb + (size_t)m0 * IN_F + (kk + BK),
                                (unsigned)(size_t)&Xs[nxt][0]);
            __builtin_prefetch(packed + (size_t)(n0 + row) * (IN_F / 4)
                                      + ((kk + 2 * BK) >> 2) + half * 8, 0, 1);
        }

        const __bf16* ap = (const __bf16*)&Xs[cur][0];
        const __bf16* bp = (const __bf16*)&Ws[cur][0];

        // ---- two K=32 substeps per staged tile: 16 WMMAs / wave / iter ----
#pragma unroll
        for (int ks = 0; ks < 2; ++ks) {
            bf16x16 afrag[2];
#pragma unroll
            for (int tm = 0; tm < 2; ++tm) {
                const int m = wm * 32 + tm * 16 + ml;
                bf16x8 lo = *(const bf16x8*)(ap + m * BK + ks * 32 + kb);
                bf16x8 hi = *(const bf16x8*)(ap + m * BK + ks * 32 + 8 + kb);
                afrag[tm] = __builtin_shufflevector(lo, hi,
                    0, 1, 2, 3, 4, 5, 6, 7, 8, 9, 10, 11, 12, 13, 14, 15);
            }

            bf16x16 bfrag[4];
#pragma unroll
            for (int tn = 0; tn < 4; ++tn) {
                const int n = wn * 64 + tn * 16 + ml;
                bfrag[tn] = *(const bf16x16*)(bp + n * BK + ks * 32 + h * 16);
            }

#pragma unroll
            for (int tm = 0; tm < 2; ++tm)
#pragma unroll
                for (int tn = 0; tn < 4; ++tn)
                    acc[tm][tn] = __builtin_amdgcn_wmma_f32_16x16x32_bf16(
                        false, afrag[tm], false, bfrag[tn],
                        (short)0, acc[tm][tn], false, false);
        }

        // TDM of next tile must land before anyone reads it after the barrier
        if (wave == 0)
            __builtin_amdgcn_s_wait_tensorcnt(0);
        __syncthreads();
        cur = nxt;
    }

    // ---- epilogue: y = acc * alpha + bias ; C/D layout m = r + 8*(lane>=16) ----
    const float alpha = *alpha_p;
#pragma unroll
    for (int tm = 0; tm < 2; ++tm) {
        const int mg = m0 + wm * 32 + tm * 16 + h * 8;
#pragma unroll
        for (int tn = 0; tn < 4; ++tn) {
            const int ng = n0 + wn * 64 + tn * 16 + ml;
            const float bv = bias[ng];
            float* op = out + (size_t)mg * OUT_F + ng;
#pragma unroll
            for (int r = 0; r < 8; ++r)
                op[(size_t)r * OUT_F] = acc[tm][tn][r] * alpha + bv;
        }
    }
}

extern "C" void kernel_launch(void* const* d_in, const int* in_sizes, int n_in,
                              void* d_out, int out_size, void* d_ws, size_t ws_size,
                              hipStream_t stream) {
    const float*         x      = (const float*)d_in[0];
    const unsigned char* packed = (const unsigned char*)d_in[1];
    const float*         alpha  = (const float*)d_in[2];
    const float*         bias   = (const float*)d_in[3];
    float* out = (float*)d_out;

    unsigned short* xb = (unsigned short*)d_ws;   // 4096*4096 bf16 = 32 MB

    // pass 1: convert x to bf16 once
    x_to_bf16_kernel<<<(NTOK * (size_t)IN_F) / (256 * 16), 256, 0, stream>>>(x, xb);

    // pass 2: WMMA GEMM
    dim3 grid(OUT_F / BN, NTOK / BM);
    ternary_wmma_bf16_kernel<<<grid, 256, 0, stream>>>(xb, packed, alpha, bias, out);
}
